// Loss_83794811945536
// MI455X (gfx1250) — matrix operand order repaired and checked by reading, not compiled, not run on hardware
//
#include <hip/hip_runtime.h>
#include <hip/hip_bf16.h>

typedef __attribute__((ext_vector_type(16))) __bf16       v16bf;
typedef __attribute__((ext_vector_type(8)))  __bf16       v8bf;
typedef __attribute__((ext_vector_type(8)))  float        v8f;
typedef __attribute__((ext_vector_type(4)))  unsigned int u32x4;
typedef __attribute__((ext_vector_type(4)))  int          i32x4;
typedef __attribute__((ext_vector_type(8)))  int          i32x8;

#define N_SAMP      4096
#define DIM         512
#define LAYERS      3
#define ALPHA_C     0.5f
#define BLK         128
#define KC          32
#define PITCH       40                    // bf16 elems per LDS row (32 + 8 pad = 80B)
#define NB          (N_SAMP / BLK)        // 32 block-rows
#define TRI_BLOCKS  (NB * (NB + 1) / 2)   // 528 upper-tri blocks per layer
#define AE_ELEMS    (N_SAMP * 784)        // 3211264
#define AE_VEC4     (AE_ELEMS / 4)        // 802816
#define NCHUNK      (DIM / KC)            // 16

// ---------------- init: zero accumulators + counts ----------------
__global__ void init_ws(float* wsf, int* counts) {
    int t = threadIdx.x;
    if (t < 2)  wsf[t]   = 0.0f;
    if (t < 16) counts[t] = 0;
}

// ---------------- bincount over labels ----------------
__global__ void bincount_k(const int* __restrict__ y, int* counts) {
    int i = blockIdx.x * blockDim.x + threadIdx.x;
    if (i < N_SAMP) atomicAdd(&counts[y[i]], 1);
}

// ---------------- per-(layer,sample) meta: (norm, weight, class, 0) ----------------
__global__ void meta_k(const float* __restrict__ emb, const int* __restrict__ y,
                       const int* __restrict__ counts, float4* __restrict__ meta) {
    int gid = blockIdx.x * blockDim.x + threadIdx.x;   // 0 .. 3*4096-1
    int l = gid >> 12;
    int i = gid & (N_SAMP - 1);
    if (l >= LAYERS) return;
    const float* base = emb + (size_t)l * DIM * N_SAMP + i;
    float nsq = 0.0f;
#pragma unroll 8
    for (int k = 0; k < DIM; ++k) {
        float v = base[(size_t)k * N_SAMP];
        nsq += v * v;
    }
    int   cls = y[i];
    float w   = 1.0f / (float)counts[cls];
    meta[(size_t)l * N_SAMP + i] = make_float4(sqrtf(nsq), w, (float)cls, 0.0f);
}

// ---------------- autoencoder MSE partial sums ----------------
__global__ void ae_k(const float* __restrict__ X, const float* __restrict__ X2,
                     float* ae_acc) {
    int i = blockIdx.x * blockDim.x + threadIdx.x;
    float s = 0.0f;
    if (i < AE_VEC4) {
        float4 a = ((const float4*)X)[i];
        float4 b = ((const float4*)X2)[i];
        float dx = a.x - b.x, dy = a.y - b.y, dz = a.z - b.z, dw = a.w - b.w;
        s = dx * dx + dy * dy + dz * dz + dw * dw;
    }
#pragma unroll
    for (int off = 16; off; off >>= 1) s += __shfl_xor(s, off, 32);
    if ((threadIdx.x & 31) == 0) atomicAdd(ae_acc, s);
}

// ---------------- one-time f32 -> bf16(hi,lo) split + transpose to [i][k] ----------------
// Tile 64(i) x 64(k); output rows have k contiguous -> TDM/LDS friendly.
__global__ void conv_k(const float* __restrict__ emb,
                       unsigned* __restrict__ hi, unsigned* __restrict__ lo) {
    __shared__ __bf16 th[64][72];
    __shared__ __bf16 tl[64][72];
    int b  = blockIdx.x;            // 3 layers * 512 tiles
    int l  = b >> 9;
    int t  = b & 511;
    int it = t >> 3;                // 0..63 i-tile
    int kt = t & 7;                 // 0..7  k-tile
    int i0 = it * 64, k0 = kt * 64;
    const float* base = emb + (size_t)l * DIM * N_SAMP;
    for (int e = threadIdx.x; e < 64 * 64; e += 256) {
        int k = e >> 6, i = e & 63;
        float v = base[(size_t)(k0 + k) * N_SAMP + i0 + i];
        __bf16 h = (__bf16)v;
        th[i][k] = h;
        tl[i][k] = (__bf16)(v - (float)h);
    }
    __syncthreads();
    for (int e = threadIdx.x; e < 64 * 32; e += 256) {
        int i = e >> 5, kk = e & 31;
        union { __bf16 b2[2]; unsigned u; } ph, pl;
        ph.b2[0] = th[i][2 * kk]; ph.b2[1] = th[i][2 * kk + 1];
        pl.b2[0] = tl[i][2 * kk]; pl.b2[1] = tl[i][2 * kk + 1];
        size_t row = (size_t)l * N_SAMP + i0 + i;
        hi[(row * DIM + (size_t)k0) / 2 + kk] = ph.u;
        lo[(row * DIM + (size_t)k0) / 2 + kk] = pl.u;
    }
}

// ---------------- shared WMMA step: frag loads (ISA 7.12.2) + bf16x3 ----------------
__device__ __forceinline__ void mma_step(const __bf16 (*__restrict__ Ahi)[PITCH],
                                         const __bf16 (*__restrict__ Alo)[PITCH],
                                         const __bf16 (*__restrict__ Bhi)[PITCH],
                                         const __bf16 (*__restrict__ Blo)[PITCH],
                                         v8f acc[4][2], int wy, int wx, int half, int l15) {
    v16bf afh[4], afl[4], bfh[2], bfl[2];
#pragma unroll
    for (int r = 0; r < 4; ++r) {
        int row = wy * 64 + r * 16 + l15;
        const __bf16* ph = &Ahi[row][half * 8];   // K {0..7}/{8..15}
        const __bf16* pl = &Alo[row][half * 8];
        v8bf h0 = *(const v8bf*)ph;
        v8bf h1 = *(const v8bf*)(ph + 16);        // K {16..23}/{24..31}
        v8bf q0 = *(const v8bf*)pl;
        v8bf q1 = *(const v8bf*)(pl + 16);
        afh[r] = __builtin_shufflevector(h0, h1, 0,1,2,3,4,5,6,7,8,9,10,11,12,13,14,15);
        afl[r] = __builtin_shufflevector(q0, q1, 0,1,2,3,4,5,6,7,8,9,10,11,12,13,14,15);
    }
#pragma unroll
    for (int c = 0; c < 2; ++c) {
        int col = wx * 32 + c * 16 + l15;
        const __bf16* ph = &Bhi[col][half * 8];   // K {0..7,16..23}/{8..15,24..31}
        const __bf16* pl = &Blo[col][half * 8];
        v8bf h0 = *(const v8bf*)ph;
        v8bf h1 = *(const v8bf*)(ph + 16);
        v8bf q0 = *(const v8bf*)pl;
        v8bf q1 = *(const v8bf*)(pl + 16);
        bfh[c] = __builtin_shufflevector(h0, h1, 0,1,2,3,4,5,6,7,8,9,10,11,12,13,14,15);
        bfl[c] = __builtin_shufflevector(q0, q1, 0,1,2,3,4,5,6,7,8,9,10,11,12,13,14,15);
    }
#pragma unroll
    for (int r = 0; r < 4; ++r)
#pragma unroll
        for (int c = 0; c < 2; ++c) {
            acc[r][c] = __builtin_amdgcn_wmma_f32_16x16x32_bf16(
                false, afl[r], false, bfh[c], (short)0, acc[r][c], false, false);
            acc[r][c] = __builtin_amdgcn_wmma_f32_16x16x32_bf16(
                false, afh[r], false, bfl[c], (short)0, acc[r][c], false, false);
            acc[r][c] = __builtin_amdgcn_wmma_f32_16x16x32_bf16(
                false, afh[r], false, bfh[c], (short)0, acc[r][c], false, false);
        }
}

// ---------------- shared epilogue: cluster loss over i<j ----------------
__device__ __forceinline__ float epilogue_sum(const v8f acc[4][2],
                                              const float4* __restrict__ metaL,
                                              int i0, int j0, int wy, int wx,
                                              int half, int l15) {
    float lsum = 0.0f;
#pragma unroll
    for (int r = 0; r < 4; ++r) {
#pragma unroll
        for (int c = 0; c < 2; ++c) {
            int ti0 = i0 + wy * 64 + r * 16;
            int tj0 = j0 + wx * 32 + c * 16;
            if (ti0 > tj0 + 15) continue;       // tile fully below diagonal
            int j = tj0 + l15;
            float4 mj = metaL[j];
#pragma unroll
            for (int q = 0; q < 8; ++q) {
                int i = ti0 + half * 8 + q;     // C-layout: vgpr q, lane-half -> M
                float  g  = acc[r][c][q];
                float4 mi = metaL[i];
                float s   = mi.x * mj.x;
                float wgt = mi.y * mj.y;
                float v   = (mi.z == mj.z) ? (s - g) : fmaxf(g - ALPHA_C * s, 0.0f);
                lsum += (i < j) ? wgt * v : 0.0f;
            }
        }
    }
    return lsum;
}

// unrank (bi, bj) with bi <= bj over NB x NB upper triangle
__device__ __forceinline__ void unrank_tri(int t, int& bi, int& bj) {
    bi = 0;
    int rem = NB;
    while (t >= rem) { t -= rem; --rem; ++bi; }
    bj = bi + t;
}

// ---------------- TDM issue: D# descriptor, 2D tile KCx128, HW LDS padding ----------------
__device__ __forceinline__ void tdm_load(unsigned lds_addr, unsigned long long gaddr,
                                         i32x8 g1) {
    u32x4 g0 = { 1u,                                  // count=1 valid descriptor
                 lds_addr,                            // LDS byte address
                 (unsigned)gaddr,                     // global_addr[31:0]
                 (unsigned)(gaddr >> 32) | (2u << 30) // global_addr[56:32] | type=2
    };
    const i32x4 z4 = {0, 0, 0, 0};
#if __clang_major__ >= 23
    const i32x8 z8 = {0, 0, 0, 0, 0, 0, 0, 0};
    __builtin_amdgcn_tensor_load_to_lds(g0, g1, z4, z4, z8, 0);
#else
    __builtin_amdgcn_tensor_load_to_lds(g0, g1, z4, z4, 0);
#endif
}

// ---------------- main Gram kernel, TDM double-buffered ----------------
__global__ void __launch_bounds__(256)
gram_tdm_k(const void* __restrict__ hiP, const void* __restrict__ loP,
           const float4* __restrict__ meta, float* ms_acc) {
    __shared__ __bf16 lbuf[2][4][BLK][PITCH];       // [buf][Ahi,Alo,Bhi,Blo]

    int blk   = blockIdx.x;
    int layer = blk / TRI_BLOCKS;
    int bi, bj;
    unrank_tri(blk % TRI_BLOCKS, bi, bj);
    const int i0 = bi * BLK, j0 = bj * BLK;

    const int tid  = threadIdx.x;
    const int lane = tid & 31;
    const int wave = tid >> 5;
    const int wy   = wave >> 2;
    const int wx   = wave & 3;
    const int half = lane >> 4;
    const int l15  = lane & 15;

    v8f acc[4][2];
#pragma unroll
    for (int r = 0; r < 4; ++r)
#pragma unroll
        for (int c = 0; c < 2; ++c) acc[r][c] = v8f{};

    const unsigned long long hb = (unsigned long long)(uintptr_t)hiP;
    const unsigned long long lb = (unsigned long long)(uintptr_t)loP;
    const unsigned long long arow = ((unsigned long long)(layer * N_SAMP + i0)) * DIM * 2ull;
    const unsigned long long brow = ((unsigned long long)(layer * N_SAMP + j0)) * DIM * 2ull;

    // D# group1: data_size=2B | pad_enable | pad_interval=16DW | pad_amount=4DW ;
    // tensor_dim0=512, tensor_dim1=3*4096, tile=32x128, strides 512 (elements).
    i32x8 g1 = { (int)((1u << 16) | (1u << 20) | (3u << 22) | (3u << 25)),
                 (int)(512u << 16),                            // tensor_dim0[15:0]
                 (int)((12288u & 0xffffu) << 16),              // dim0[31:16]|dim1[15:0]
                 (int)((12288u >> 16) | (32u << 16)),          // dim1[31:16]|tile_dim0
                 (int)128,                                     // tile_dim1 | tile_dim2=0
                 (int)512,                                     // dim0_stride[31:0]
                 (int)(512u << 16),                            // stride hi | dim1_stride lo
                 0 };

    const int NC = NCHUNK;
    auto issue4 = [&](int b, int kc) {
        unsigned long long ko = (unsigned long long)kc * 2ull;
        tdm_load((unsigned)(uintptr_t)&lbuf[b][0][0][0], hb + arow + ko, g1);
        tdm_load((unsigned)(uintptr_t)&lbuf[b][1][0][0], lb + arow + ko, g1);
        tdm_load((unsigned)(uintptr_t)&lbuf[b][2][0][0], hb + brow + ko, g1);
        tdm_load((unsigned)(uintptr_t)&lbuf[b][3][0][0], lb + brow + ko, g1);
    };

    if (wave == 0) issue4(0, 0);
    for (int c = 0; c < NC; ++c) {
        int cur = c & 1;
        if (wave == 0) {
            if (c + 1 < NC) {
                issue4(1 - cur, (c + 1) * KC);               // prefetch next chunk
                __builtin_amdgcn_s_wait_tensorcnt((short)4); // chunk c landed (in-order)
            } else {
                __builtin_amdgcn_s_wait_tensorcnt((short)0);
            }
        }
        __syncthreads();
        mma_step(lbuf[cur][0], lbuf[cur][1], lbuf[cur][2], lbuf[cur][3],
                 acc, wy, wx, half, l15);
        __syncthreads();   // readers done: buffer reusable by next prefetch
    }

    float lsum = epilogue_sum(acc, meta + (size_t)layer * N_SAMP, i0, j0, wy, wx, half, l15);
#pragma unroll
    for (int off = 16; off; off >>= 1) lsum += __shfl_xor(lsum, off, 32);
    if (lane == 0) atomicAdd(ms_acc, lsum);
}

// ---------------- fallback Gram kernel (in-kernel conversion staging) ----------------
__global__ void __launch_bounds__(256)
gram_fb_k(const float* __restrict__ emb, const float4* __restrict__ meta,
          float* ms_acc) {
    __shared__ __bf16 lbuf[4][BLK][PITCH];          // Ahi, Alo, Bhi, Blo

    int blk   = blockIdx.x;
    int layer = blk / TRI_BLOCKS;
    int bi, bj;
    unrank_tri(blk % TRI_BLOCKS, bi, bj);
    const int i0 = bi * BLK, j0 = bj * BLK;
    const float* Abase = emb + (size_t)layer * DIM * N_SAMP;

    const int tid  = threadIdx.x;
    const int lane = tid & 31;
    const int wave = tid >> 5;
    const int wy   = wave >> 2;
    const int wx   = wave & 3;
    const int half = lane >> 4;
    const int l15  = lane & 15;

    v8f acc[4][2];
#pragma unroll
    for (int r = 0; r < 4; ++r)
#pragma unroll
        for (int c = 0; c < 2; ++c) acc[r][c] = v8f{};

    for (int kc = 0; kc < DIM; kc += KC) {
        for (int e = tid; e < (KC * BLK / 4); e += 256) {
            int k    = e >> 5;
            int c4   = e & 31;
            int icol = c4 * 4;
            const float* rowp = Abase + (size_t)(kc + k) * N_SAMP;
            float4 va = *(const float4*)(rowp + i0 + icol);
            float4 vb = *(const float4*)(rowp + j0 + icol);
            float av[4] = {va.x, va.y, va.z, va.w};
            float bv[4] = {vb.x, vb.y, vb.z, vb.w};
#pragma unroll
            for (int q = 0; q < 4; ++q) {
                __bf16 h = (__bf16)av[q];
                lbuf[0][icol + q][k] = h;
                lbuf[1][icol + q][k] = (__bf16)(av[q] - (float)h);
                __bf16 g = (__bf16)bv[q];
                lbuf[2][icol + q][k] = g;
                lbuf[3][icol + q][k] = (__bf16)(bv[q] - (float)g);
            }
        }
        __syncthreads();
        mma_step(lbuf[0], lbuf[1], lbuf[2], lbuf[3], acc, wy, wx, half, l15);
        __syncthreads();
    }

    float lsum = epilogue_sum(acc, meta + (size_t)layer * N_SAMP, i0, j0, wy, wx, half, l15);
#pragma unroll
    for (int off = 16; off; off >>= 1) lsum += __shfl_xor(lsum, off, 32);
    if (lane == 0) atomicAdd(ms_acc, lsum);
}

// ---------------- finalize: compose the 3 outputs ----------------
__global__ void finalize_k(const float* __restrict__ wsf, float* __restrict__ out) {
    float ms = wsf[0] / (float)N_SAMP;
    float ae = wsf[1] / (float)AE_ELEMS;
    out[0] = ms + ae;
    out[1] = ms;
    out[2] = ae;
}

extern "C" void kernel_launch(void* const* d_in, const int* in_sizes, int n_in,
                              void* d_out, int out_size, void* d_ws, size_t ws_size,
                              hipStream_t stream) {
    const float* X   = (const float*)d_in[0];
    const float* X_  = (const float*)d_in[1];
    const float* emb = (const float*)d_in[2];
    const int*   y   = (const int*)d_in[3];
    float* out = (float*)d_out;

    char*   ws     = (char*)d_ws;
    float*  wsf    = (float*)ws;            // [0]=ms acc, [1]=ae acc
    int*    counts = (int*)(ws + 64);       // 16 ints
    float4* meta   = (float4*)(ws + 256);   // [3][4096] float4 ~192KB

    const size_t META_BYTES = (size_t)LAYERS * N_SAMP * sizeof(float4);
    const size_t HI_OFF     = (256 + META_BYTES + 255) & ~(size_t)255;
    const size_t HL_BYTES   = (size_t)LAYERS * N_SAMP * DIM * 2;   // 12 MB each
    const size_t NEED       = HI_OFF + 2 * HL_BYTES;               // ~25.4 MB
    const bool   use_tdm    = (ws_size >= NEED);

    init_ws   <<<1, 32, 0, stream>>>(wsf, counts);
    bincount_k<<<N_SAMP / 256, 256, 0, stream>>>(y, counts);
    meta_k    <<<(LAYERS * N_SAMP) / 256, 256, 0, stream>>>(emb, y, counts, meta);
    ae_k      <<<AE_VEC4 / 256, 256, 0, stream>>>(X, X_, wsf + 1);

    if (use_tdm) {
        unsigned* hi = (unsigned*)(ws + HI_OFF);
        unsigned* lo = (unsigned*)(ws + HI_OFF + HL_BYTES);
        conv_k    <<<LAYERS * 512, 256, 0, stream>>>(emb, hi, lo);
        gram_tdm_k<<<LAYERS * TRI_BLOCKS, 256, 0, stream>>>((const void*)hi, (const void*)lo,
                                                            meta, wsf);
    } else {
        gram_fb_k <<<LAYERS * TRI_BLOCKS, 256, 0, stream>>>(emb, meta, wsf);
    }
    finalize_k<<<1, 1, 0, stream>>>(wsf, out);
}
